// ShadowChebKANEncoder_30236569764239
// MI455X (gfx1250) — matrix operand order repaired and checked by reading, compiled, not run on hardware
//
#include <hip/hip_runtime.h>
#include <cstdint>

typedef __attribute__((ext_vector_type(16))) __bf16 v16bf;
typedef __attribute__((ext_vector_type(8)))  __bf16 v8bf;
typedef __attribute__((ext_vector_type(2)))  __bf16 v2bf;
typedef __attribute__((ext_vector_type(8)))  float  v8f;
typedef __attribute__((ext_vector_type(4)))  float  v4f;
typedef __attribute__((ext_vector_type(2)))  float  v2f;

#define N_ROWS   8192
#define D_IN     1024
#define D_MODEL  1024
#define K_ORD    9          // weights[0..8]
#define BM       128
#define BN       128
#define BK       32
#define XS_STRIDE 36        // f32 per x-row in LDS (144B row, 16B aligned)
#define BW_STRIDE 40        // bf16 per weight-column in LDS (80B stride)
#define AW_STRIDE 40        // bf16 per T_k-row in LDS (80B stride)
#define NTHREADS 256        // 8 waves; per-wave output tile 32x64

static __device__ __forceinline__ v8f splat8(float s) {
    v8f r;
#pragma unroll
    for (int e = 0; e < 8; ++e) r[e] = s;
    return r;
}

__global__ __launch_bounds__(NTHREADS, 1)
void cheb_kan_wmma(const float* __restrict__ x,
                   const float* __restrict__ w,
                   float* __restrict__ out)
{
    __shared__ __align__(16) float  xs[2][BM * XS_STRIDE];  // x chunks (async-staged), dbl-buf
    __shared__ __align__(16) __bf16 bw[2][BN * BW_STRIDE];  // W_k chunk, bf16 transposed [n][k], dbl-buf
    __shared__ __align__(16) __bf16 aw[2][BM * AW_STRIDE];  // T_k chunk, bf16 row-major [m][k], dbl-buf

    const int tid      = threadIdx.x;
    const int lane     = tid & 31;
    const int wave     = tid >> 5;
    const int waveM    = wave >> 1;       // 0..3 -> rows 32*waveM
    const int waveN    = wave & 1;        // 0..1 -> cols 64*waveN
    const int laneRow  = lane & 15;
    const int laneHalf = lane >> 4;
    const int kbaseA   = laneHalf * 8;    // A frag: K in {base..base+7} U {base+16..base+23}
    const int kbaseB   = laneHalf * 8;    // B frag: K in {base..base+7} U {base+16..base+23}

    const int blockM = blockIdx.y * BM;
    const int blockN = blockIdx.x * BN;

    // cooperative Chebyshev distribution: 256 threads = 128 rows x 2 column-hexadectets
    const int rr  = tid >> 1;             // row 0..127 of the x tile
    const int dox = (tid & 1) * 16;       // column group 0 or 16 (two octets each)

    // weight staging geometry: 512 units = 16 (d-pairs) x 32 (m-quads); 2 units/thread
    const int wdd = ((tid & 255) >> 5) * 2;  // even reduction row 0..14 (+16 for unit 1)
    const int wmm = (tid & 31) * 4;          // output col quad 0..124

    v8f acc[2][4];
#pragma unroll
    for (int t = 0; t < 2; ++t)
#pragma unroll
        for (int j = 0; j < 4; ++j)
            acc[t][j] = (v8f){0.f,0.f,0.f,0.f,0.f,0.f,0.f,0.f};

    // ---- issue async x-copy for chunk 0 ----
#pragma unroll
    for (int i = 0; i < 4; ++i) {
        int seg = tid + i * NTHREADS;     // 1024 segments of 16B
        int row = seg >> 3;
        int c4  = (seg & 7) << 2;
        unsigned ldsOff = (unsigned)(uintptr_t)(&xs[0][row * XS_STRIDE + c4]);
        unsigned long long ga =
            (unsigned long long)(uintptr_t)(x + (size_t)(blockM + row) * D_IN + c4);
        asm volatile("global_load_async_to_lds_b128 %0, %1, off"
                     :: "v"(ldsOff), "v"(ga) : "memory");
    }

    for (int ci = 0; ci < D_IN / BK; ++ci) {
        const int d0 = ci * BK;
        asm volatile("s_wait_asynccnt 0x0" ::: "memory");
        __syncthreads();                  // x buf[ci&1] visible to all waves

        // ---- prefetch next chunk's x tile (overlaps this chunk's compute) ----
        if (ci + 1 < D_IN / BK) {
#pragma unroll
            for (int i = 0; i < 4; ++i) {
                int seg = tid + i * NTHREADS;
                int row = seg >> 3;
                int c4  = (seg & 7) << 2;
                unsigned ldsOff =
                    (unsigned)(uintptr_t)(&xs[(ci + 1) & 1][row * XS_STRIDE + c4]);
                unsigned long long ga = (unsigned long long)(uintptr_t)
                    (x + (size_t)(blockM + row) * D_IN + d0 + BK + c4);
                asm volatile("global_load_async_to_lds_b128 %0, %1, off"
                             :: "v"(ldsOff), "v"(ga) : "memory");
            }
        }

        // ---- this thread's 16 x-values for the cooperative recurrence ----
        v8f x2[2], p1[2], p2[2];          // 2*x and Chebyshev T_{k-1}, T_{k-2} (f32)
#pragma unroll
        for (int g = 0; g < 2; ++g) {
            const float* xp = &xs[ci & 1][rr * XS_STRIDE + dox + g * 8];
            v4f xa = *(const v4f*)(xp);
            v4f xb = *(const v4f*)(xp + 4);
            v8f xt = __builtin_shufflevector(xa, xb, 0,1,2,3,4,5,6,7);
            x2[g] = xt + xt;
        }

        // ---- prefetch W_0 chunk into registers (2 units: wdd and wdd+16) ----
        const float* wbase = w + (size_t)d0 * D_MODEL + blockN;
        v4f wr[2][2];
#pragma unroll
        for (int u = 0; u < 2; ++u) {
            int dd = wdd + u * 16;
            wr[u][0] = *(const v4f*)(wbase + (size_t)dd * D_MODEL + wmm);
            wr[u][1] = *(const v4f*)(wbase + (size_t)(dd + 1) * D_MODEL + wmm);
        }

#pragma unroll
        for (int k = 0; k < K_ORD; ++k) {
            // ---- cooperative T_k: 8 pk-fma + 2 b128 stores per thread ----
#pragma unroll
            for (int g = 0; g < 2; ++g) {
                v8f cur;
                if (k == 0)      cur = splat8(1.0f);
                else if (k == 1) cur = x2[g] * 0.5f;
                else             cur = __builtin_elementwise_fma(x2[g], p1[g], -p2[g]);
                if (k > 0) p2[g] = p1[g];
                p1[g] = cur;
                *(v8bf*)(&aw[k & 1][rr * AW_STRIDE + dox + g * 8]) =
                    __builtin_convertvector(cur, v8bf);
            }

            // ---- commit prefetched W_k regs -> bf16 transposed LDS (buf k&1) ----
#pragma unroll
            for (int u = 0; u < 2; ++u) {
                int dd = wdd + u * 16;
#pragma unroll
                for (int q = 0; q < 4; ++q) {
                    v2f pr = (v2f){wr[u][0][q], wr[u][1][q]};
                    *(v2bf*)(&bw[k & 1][(wmm + q) * BW_STRIDE + dd]) =
                        __builtin_convertvector(pr, v2bf);
                }
            }
            // ---- issue global loads for W_{k+1} (overlap with WMMAs below) ----
            if (k + 1 < K_ORD) {
                const float* wn = w + ((size_t)(k + 1) * D_IN + d0) * D_MODEL + blockN;
#pragma unroll
                for (int u = 0; u < 2; ++u) {
                    int dd = wdd + u * 16;
                    wr[u][0] = *(const v4f*)(wn + (size_t)dd * D_MODEL + wmm);
                    wr[u][1] = *(const v4f*)(wn + (size_t)(dd + 1) * D_MODEL + wmm);
                }
            }
            __syncthreads();              // aw/bw[k&1] writes visible; k-2 readers fenced at k-1

            // ---- A fragments from LDS (reused across 4 N-tiles) ----
            v16bf afrag[2];
#pragma unroll
            for (int t = 0; t < 2; ++t) {
                int r = waveM * 32 + t * 16 + laneRow;
                const __bf16* ap = &aw[k & 1][r * AW_STRIDE + kbaseA];
                v8bf alo = *(const v8bf*)(ap);        // K = base+0..7
                v8bf ahi = *(const v8bf*)(ap + 16);   // K = base+16..23
                afrag[t] = __builtin_shufflevector(
                    alo, ahi, 0,1,2,3,4,5,6,7,8,9,10,11,12,13,14,15);
            }

            // ---- B fragments from LDS + 8 WMMAs ----
#pragma unroll
            for (int j = 0; j < 4; ++j) {
                int c = waveN * 64 + j * 16 + laneRow;
                const __bf16* bp = &bw[k & 1][c * BW_STRIDE + kbaseB];
                v8bf blo = *(const v8bf*)(bp);        // K = base+0..7
                v8bf bhi = *(const v8bf*)(bp + 16);   // K = base+16..23
                v16bf bfrag = __builtin_shufflevector(
                    blo, bhi, 0,1,2,3,4,5,6,7,8,9,10,11,12,13,14,15);
#pragma unroll
                for (int t = 0; t < 2; ++t)
                    acc[t][j] = __builtin_amdgcn_wmma_f32_16x16x32_bf16(
                        false, afrag[t], false, bfrag,
                        (short)0, acc[t][j], false, false);
            }
        }
        __syncthreads();                  // chunk's readers done before next chunk's writes
    }

    // ---- store 128x128 f32 tile (C layout: VGPR v -> M=v+8*laneHalf, N=laneRow) ----
#pragma unroll
    for (int t = 0; t < 2; ++t)
#pragma unroll
        for (int j = 0; j < 4; ++j)
#pragma unroll
            for (int v = 0; v < 8; ++v) {
                int row = blockM + waveM * 32 + t * 16 + laneHalf * 8 + v;
                int col = blockN + waveN * 64 + j * 16 + laneRow;
                out[(size_t)row * D_MODEL + col] = acc[t][j][v];
            }
}

extern "C" void kernel_launch(void* const* d_in, const int* in_sizes, int n_in,
                              void* d_out, int out_size, void* d_ws, size_t ws_size,
                              hipStream_t stream) {
    const float* x  = (const float*)d_in[0];   // [8192, 1024] f32
    const float* w  = (const float*)d_in[1];   // [9, 1024, 1024] f32
    float* out      = (float*)d_out;           // [8192, 1024] f32
    dim3 grid(D_MODEL / BN, N_ROWS / BM);      // (8, 64) = 512 workgroups
    cheb_kan_wmma<<<grid, dim3(NTHREADS), 0, stream>>>(x, w, out);
}
